// MultiModalGraphBlock_47399259079066
// MI455X (gfx1250) — matrix operand order repaired and checked by reading, compiled, not run on hardware
//
#include <hip/hip_runtime.h>
#include <hip/hip_bf16.h>
#include <stdint.h>

#define DIM   128
#define HEADS 8
#define HC    16
#define LEAKY 0.2f

typedef __attribute__((ext_vector_type(16))) _Float16 v16h;
typedef __attribute__((ext_vector_type(8)))  _Float16 v8h;
typedef __attribute__((ext_vector_type(8)))  float    v8f;

// ---------------------------------------------------------------- utilities

__device__ __forceinline__ void atomic_max_f32(float* addr, float val) {
  // monotone int-bits trick: works for mixed signs given -inf init
  if (val >= 0.0f) atomicMax((int*)addr, __float_as_int(val));
  else             atomicMin((unsigned int*)addr, __float_as_uint(val));
}

__device__ __forceinline__ float wave_reduce_add(float v) {
#pragma unroll
  for (int o = 16; o > 0; o >>= 1) v += __shfl_xor(v, o, 32);
  return v;
}

// ---------------------------------------------------------------- init

__global__ void k_init(float* m_gat, float* s_gat, float* m_tr, float* s_tr,
                       float* deg, float* gat_acc, float* tr_acc, float* nbr,
                       int n) {
  int i = blockIdx.x * blockDim.x + threadIdx.x;
  if (i < n * HEADS) {
    m_gat[i] = -__builtin_inff();
    m_tr[i]  = -__builtin_inff();
    s_gat[i] = 0.0f;
    s_tr[i]  = 0.0f;
  }
  if (i < n) deg[i] = 0.0f;
  if (i < n * DIM) { gat_acc[i] = 0.0f; tr_acc[i] = 0.0f; nbr[i] = 0.0f; }
}

// ------------------------------------------------- weight repack (f32 -> f16 Bt[N][K])

__global__ void k_prep(const float* __restrict__ gat_w,  const float* __restrict__ sage_wr,
                       const float* __restrict__ wq,     const float* __restrict__ wk,
                       const float* __restrict__ wv,     const float* __restrict__ wskip,
                       const float* __restrict__ bq,     const float* __restrict__ bk,
                       const float* __restrict__ bv,     const float* __restrict__ bskip,
                       const float* __restrict__ sage_wl, const float* __restrict__ fus_w,
                       _Float16* __restrict__ Bt6, float* __restrict__ bias6,
                       _Float16* __restrict__ Btl, _Float16* __restrict__ Btf) {
  int i = blockIdx.x * blockDim.x + threadIdx.x;
  if (i < 768 * 128) {                       // Bt6[ncol][k] = W6[k][ncol]
    int ncol = i / 128, k = i % 128;
    int seg = ncol / 128, c = ncol % 128;
    const float* W = gat_w;
    if      (seg == 1) W = sage_wr;
    else if (seg == 2) W = wq;
    else if (seg == 3) W = wk;
    else if (seg == 4) W = wv;
    else if (seg == 5) W = wskip;
    Bt6[(size_t)ncol * 128 + k] = (_Float16)W[k * 128 + c];
  }
  if (i < 768) {                             // per-segment bias vector
    int seg = i / 128, c = i % 128;
    float b = 0.0f;
    if      (seg == 2) b = bq[c];
    else if (seg == 3) b = bk[c];
    else if (seg == 4) b = bv[c];
    else if (seg == 5) b = bskip[c];
    bias6[i] = b;
  }
  if (i < 128 * 128) {                       // Btl[ncol][k] = sage_wl[k][ncol]
    int ncol = i / 128, k = i % 128;
    Btl[i] = (_Float16)sage_wl[k * 128 + ncol];
  }
  if (i < 128 * 384) {                       // Btf[ncol][k] = fus_w[k][ncol], K=384
    int ncol = i / 384, k = i % 384;
    Btf[i] = (_Float16)fus_w[k * 128 + ncol];
  }
}

// ---------------------------------------------------------------- WMMA GEMM
// C[M x ldc] = A[M x K](f32) * B[K x ncols] + bias, B given transposed as
// Bt[ncols][K] f16.  One block per 16-row M tile: A tile staged once in LDS
// as f16, A fragments kept in VGPRs; 4 waves sweep the column tiles with a
// one-deep software pipeline on the B fragments so each v_wmma overlaps the
// next B fetch.  Fully unrolled K -> K/32 v_wmma_f32_16x16x32_f16.

template <int K>
__global__ __launch_bounds__(128)
void k_wmma_gemm(const float* __restrict__ A, const _Float16* __restrict__ Bt,
                 const float* __restrict__ bias, float* __restrict__ C,
                 int M, int ncols, int ldc) {
  constexpr int KF = K / 32;                 // k-fragments
  __shared__ _Float16 As[16 * K];            // A tile, row-major [16][K], f16

  const int tid   = threadIdx.x;
  const int lane  = tid & 31;
  const int wid   = tid >> 5;
  const int r     = lane & 15;               // row-in-tile (A) / col-in-tile (B, C)
  const int hf    = lane >> 4;               // K-half selector
  const int tileM = blockIdx.x * 16;

  // ---- cooperative stage: 16 x K f32 -> f16 into LDS (4 elems / thread / iter)
  for (int idx = tid * 4; idx < 16 * K; idx += 128 * 4) {
    int row = idx / K, col = idx % K;
    int grow = tileM + row;
    if (grow >= M) grow = M - 1;
    const float* src = A + (size_t)grow * K + col;
    _Float16* dst = As + idx;
    dst[0] = (_Float16)src[0];
    dst[1] = (_Float16)src[1];
    dst[2] = (_Float16)src[2];
    dst[3] = (_Float16)src[3];
  }
  __syncthreads();

  // ---- A fragments, resident in VGPRs for the whole column sweep
  const _Float16* ar = As + r * K;
  v16h afrag[KF];
#pragma unroll
  for (int kf = 0; kf < KF; ++kf) {
    v8h a0 = *(const v8h*)(ar + kf * 32 + hf * 8);
    v8h a1 = *(const v8h*)(ar + kf * 32 + 16 + hf * 8);
    v16h a;
#pragma unroll
    for (int j = 0; j < 8; ++j) { a[j] = a0[j]; a[8 + j] = a1[j]; }
    afrag[kf] = a;
  }

  // ---- sweep column tiles (wave w owns tiles w, w+4, w+8, ...)
  for (int tn = wid * 16; tn < ncols; tn += 64) {
    const _Float16* bcol = Bt + (size_t)(tn + r) * K;
    if (tn + 64 < ncols)                     // gfx1250 global_prefetch_b8
      __builtin_prefetch(Bt + (size_t)(tn + 64 + r) * K, 0, 1);
    const float bv = bias ? bias[tn + r] : 0.0f;   // hoisted above the MMAs

    v8f acc = {0.f, 0.f, 0.f, 0.f, 0.f, 0.f, 0.f, 0.f};
    v16h bcur = __builtin_shufflevector(*(const v8h*)(bcol + hf * 8), *(const v8h*)(bcol + 16 + hf * 8), 0, 1, 2, 3, 4, 5, 6, 7, 8, 9, 10, 11, 12, 13, 14, 15);    // prologue fetch
#pragma unroll
    for (int kf = 0; kf < KF; ++kf) {
      v16h bnxt;
      if (kf + 1 < KF)                       // fetch-ahead: overlaps the wmma
        bnxt = __builtin_shufflevector(*(const v8h*)(bcol + (kf + 1) * 32 + hf * 8), *(const v8h*)(bcol + (kf + 1) * 32 + 16 + hf * 8), 0, 1, 2, 3, 4, 5, 6, 7, 8, 9, 10, 11, 12, 13, 14, 15);
      acc = __builtin_amdgcn_wmma_f32_16x16x32_f16(false, afrag[kf], false, bcur,
                                                   (short)0, acc, false, false);
      bcur = bnxt;
    }

    // ---- epilogue: full tiles take the unconditional path (M % 16 == 0 here)
    const int mb = tileM + hf * 8;           // C/D: VGPR g -> M = hf*8 + g, N = r
    if (tileM + 16 <= M) {
#pragma unroll
      for (int g = 0; g < 8; ++g)
        C[(size_t)(mb + g) * ldc + tn + r] = acc[g] + bv;
    } else {
#pragma unroll
      for (int g = 0; g < 8; ++g)
        if (mb + g < M) C[(size_t)(mb + g) * ldc + tn + r] = acc[g] + bv;
    }
  }
}

// ------------------------------------------ per-node GAT attention scalars

__global__ void k_alpha(const float* __restrict__ H,
                        const float* __restrict__ att_src, const float* __restrict__ att_dst,
                        float* __restrict__ a_src, float* __restrict__ a_dst, int n) {
  int i = blockIdx.x * blockDim.x + threadIdx.x;
  if (i >= n * HEADS) return;
  int node = i / HEADS, h = i % HEADS;
  const float* hp = H + (size_t)node * 768 + h * HC;   // gat segment = cols 0..127
  float s = 0.0f, d = 0.0f;
#pragma unroll
  for (int c = 0; c < HC; ++c) {
    float v = hp[c];
    s += v * att_src[h * HC + c];
    d += v * att_dst[h * HC + c];
  }
  a_src[i] = s;
  a_dst[i] = d;
}

// ------------------------------------------ edge pass 1: segment max (GAT + TR)

__global__ void k_edge_max(const long long* __restrict__ ei, long long E,
                           const float* __restrict__ H,
                           const float* __restrict__ a_src, const float* __restrict__ a_dst,
                           float* __restrict__ m_gat, float* __restrict__ m_tr) {
  long long i = (long long)blockIdx.x * blockDim.x + threadIdx.x;
  if (i >= E * HEADS) return;
  long long e = i / HEADS;
  int h = (int)(i % HEADS);
  long long s = ei[e], d = ei[E + e];

  float z = a_src[s * HEADS + h] + a_dst[d * HEADS + h];
  z = (z > 0.0f) ? z : LEAKY * z;
  atomic_max_f32(&m_gat[d * HEADS + h], z);

  const float4* q4 = (const float4*)(H + (size_t)d * 768 + 256 + h * HC);
  const float4* k4 = (const float4*)(H + (size_t)s * 768 + 384 + h * HC);
  float t = 0.0f;
#pragma unroll
  for (int c = 0; c < 4; ++c) {
    float4 qv = q4[c], kv = k4[c];
    t += qv.x * kv.x + qv.y * kv.y + qv.z * kv.z + qv.w * kv.w;
  }
  t *= 0.25f;                                           // 1/sqrt(16)
  atomic_max_f32(&m_tr[d * HEADS + h], t);
}

// ------------------------------------------ GAT self-loop folded into max

__global__ void k_self_max(const float* __restrict__ a_src, const float* __restrict__ a_dst,
                           float* __restrict__ m_gat, int n) {
  int i = blockIdx.x * blockDim.x + threadIdx.x;
  if (i >= n * HEADS) return;
  float z = a_src[i] + a_dst[i];
  z = (z > 0.0f) ? z : LEAKY * z;
  m_gat[i] = fmaxf(m_gat[i], z);
}

// ------------------------------------------ edge pass 2: accumulate num/denom

__global__ void k_edge_acc(const long long* __restrict__ ei, long long E,
                           const float* __restrict__ x, const float* __restrict__ H,
                           const float* __restrict__ a_src, const float* __restrict__ a_dst,
                           const float* __restrict__ m_gat, const float* __restrict__ m_tr,
                           float* __restrict__ s_gat, float* __restrict__ s_tr,
                           float* __restrict__ gat_acc, float* __restrict__ tr_acc,
                           float* __restrict__ nbr, float* __restrict__ deg) {
  long long i = (long long)blockIdx.x * blockDim.x + threadIdx.x;
  if (i >= E * HEADS) return;
  long long e = i / HEADS;
  int h = (int)(i % HEADS);
  long long s = ei[e], d = ei[E + e];

  // GAT: unnormalized weight and weighted message
  float z = a_src[s * HEADS + h] + a_dst[d * HEADS + h];
  z = (z > 0.0f) ? z : LEAKY * z;
  float w = __expf(z - m_gat[d * HEADS + h]);
  atomicAdd(&s_gat[d * HEADS + h], w);
  const float4* h4 = (const float4*)(H + (size_t)s * 768 + h * HC);
  float* ga = gat_acc + (size_t)d * DIM + h * HC;
#pragma unroll
  for (int c = 0; c < 4; ++c) {
    float4 hv = h4[c];
    atomicAdd(&ga[c * 4 + 0], w * hv.x);
    atomicAdd(&ga[c * 4 + 1], w * hv.y);
    atomicAdd(&ga[c * 4 + 2], w * hv.z);
    atomicAdd(&ga[c * 4 + 3], w * hv.w);
  }

  // TransformerConv: q.k logit, weighted v
  const float4* q4 = (const float4*)(H + (size_t)d * 768 + 256 + h * HC);
  const float4* k4 = (const float4*)(H + (size_t)s * 768 + 384 + h * HC);
  float t = 0.0f;
#pragma unroll
  for (int c = 0; c < 4; ++c) {
    float4 qv = q4[c], kv = k4[c];
    t += qv.x * kv.x + qv.y * kv.y + qv.z * kv.z + qv.w * kv.w;
  }
  t *= 0.25f;
  float wt = __expf(t - m_tr[d * HEADS + h]);
  atomicAdd(&s_tr[d * HEADS + h], wt);
  const float4* v4 = (const float4*)(H + (size_t)s * 768 + 512 + h * HC);
  float* ta = tr_acc + (size_t)d * DIM + h * HC;
#pragma unroll
  for (int c = 0; c < 4; ++c) {
    float4 vv = v4[c];
    atomicAdd(&ta[c * 4 + 0], wt * vv.x);
    atomicAdd(&ta[c * 4 + 1], wt * vv.y);
    atomicAdd(&ta[c * 4 + 2], wt * vv.z);
    atomicAdd(&ta[c * 4 + 3], wt * vv.w);
  }

  // SAGE: plain neighbor sum + degree
  const float4* x4 = (const float4*)(x + (size_t)s * DIM + h * HC);
  float* na = nbr + (size_t)d * DIM + h * HC;
#pragma unroll
  for (int c = 0; c < 4; ++c) {
    float4 xv = x4[c];
    atomicAdd(&na[c * 4 + 0], xv.x);
    atomicAdd(&na[c * 4 + 1], xv.y);
    atomicAdd(&na[c * 4 + 2], xv.z);
    atomicAdd(&na[c * 4 + 3], xv.w);
  }
  if (h == 0) atomicAdd(&deg[d], 1.0f);
}

// ------------------------------------------ node finalize (GAT, TR, SAGE mean)

__global__ void k_node_mid(const float* __restrict__ H,
                           const float* __restrict__ a_src, const float* __restrict__ a_dst,
                           const float* __restrict__ m_gat, const float* __restrict__ s_gat,
                           const float* __restrict__ s_tr,
                           const float* __restrict__ gat_acc, const float* __restrict__ tr_acc,
                           const float* __restrict__ gat_b,
                           float* __restrict__ nbr, const float* __restrict__ deg,
                           float* __restrict__ concat, int n) {
  int i = blockIdx.x * blockDim.x + threadIdx.x;
  if (i >= n * DIM) return;
  int node = i / DIM, c = i % DIM, h = c / HC;
  size_t nh = (size_t)node * HEADS + h;

  // GAT: add self-loop term, normalize, bias
  float z = a_src[nh] + a_dst[nh];
  z = (z > 0.0f) ? z : LEAKY * z;
  float es   = __expf(z - m_gat[nh]);
  float ssum = s_gat[nh] + es;
  float gnum = gat_acc[i] + es * H[(size_t)node * 768 + c];
  concat[(size_t)node * 384 + c] = gnum / fmaxf(ssum, 1e-16f) + gat_b[c];

  // TransformerConv: normalize + skip (skip branch already has bias)
  float tr = tr_acc[i] / fmaxf(s_tr[nh], 1e-16f) + H[(size_t)node * 768 + 640 + c];
  concat[(size_t)node * 384 + 256 + c] = tr;

  // SAGE: mean (in place, feeds next GEMM)
  nbr[i] = nbr[i] / fmaxf(deg[node], 1.0f);
}

// ------------------------------------------ SAGE combine + L2 normalize

__global__ __launch_bounds__(256)
void k_sage_norm(const float* __restrict__ sage_tmp, const float* __restrict__ H,
                 float* __restrict__ concat, int n) {
  int node = (blockIdx.x * blockDim.x + threadIdx.x) >> 5;
  int lane = threadIdx.x & 31;
  if (node >= n) return;
  float v[4];
  float ss = 0.0f;
#pragma unroll
  for (int j = 0; j < 4; ++j) {
    int c = lane * 4 + j;
    float t = sage_tmp[(size_t)node * DIM + c] + H[(size_t)node * 768 + 128 + c];
    v[j] = t;
    ss += t * t;
  }
  ss = wave_reduce_add(ss);
  float inv = 1.0f / fmaxf(sqrtf(ss), 1e-12f);
#pragma unroll
  for (int j = 0; j < 4; ++j)
    concat[(size_t)node * 384 + 128 + lane * 4 + j] = v[j] * inv;
}

// ------------------------------------------ LayerNorm (wave per node)

__global__ __launch_bounds__(256)
void k_layernorm(const float* __restrict__ pre, const float* __restrict__ g,
                 const float* __restrict__ b, float* __restrict__ out, int n) {
  int node = (blockIdx.x * blockDim.x + threadIdx.x) >> 5;
  int lane = threadIdx.x & 31;
  if (node >= n) return;
  float v[4];
  float s = 0.0f;
#pragma unroll
  for (int j = 0; j < 4; ++j) {
    v[j] = pre[(size_t)node * DIM + lane * 4 + j];
    s += v[j];
  }
  s = wave_reduce_add(s);
  float mu = s * (1.0f / DIM);
  float var = 0.0f;
#pragma unroll
  for (int j = 0; j < 4; ++j) { float d = v[j] - mu; var += d * d; }
  var = wave_reduce_add(var) * (1.0f / DIM);
  float inv = rsqrtf(var + 1e-5f);
#pragma unroll
  for (int j = 0; j < 4; ++j) {
    int c = lane * 4 + j;
    out[(size_t)node * DIM + c] = (v[j] - mu) * inv * g[c] + b[c];
  }
}

// ---------------------------------------------------------------- launcher

extern "C" void kernel_launch(void* const* d_in, const int* in_sizes, int n_in,
                              void* d_out, int out_size, void* d_ws, size_t ws_size,
                              hipStream_t stream) {
  const float*     x        = (const float*)d_in[0];
  const long long* ei       = (const long long*)d_in[1];   // int64 edge_index
  const float*     gat_w    = (const float*)d_in[2];
  const float*     att_src  = (const float*)d_in[3];
  const float*     att_dst  = (const float*)d_in[4];
  const float*     gat_b    = (const float*)d_in[5];
  const float*     sage_wl  = (const float*)d_in[6];
  const float*     sage_bl  = (const float*)d_in[7];
  const float*     sage_wr  = (const float*)d_in[8];
  const float*     wq       = (const float*)d_in[9];
  const float*     bq       = (const float*)d_in[10];
  const float*     wk       = (const float*)d_in[11];
  const float*     bk       = (const float*)d_in[12];
  const float*     wv       = (const float*)d_in[13];
  const float*     bv       = (const float*)d_in[14];
  const float*     wskip    = (const float*)d_in[15];
  const float*     bskip    = (const float*)d_in[16];
  const float*     fus_w    = (const float*)d_in[17];
  const float*     fus_b    = (const float*)d_in[18];
  const float*     ln_g     = (const float*)d_in[19];
  const float*     ln_b     = (const float*)d_in[20];

  const int       n = in_sizes[0] / DIM;
  const long long E = (long long)in_sizes[1] / 2;

  // ---- workspace carve-up
  char* p = (char*)d_ws;
  auto alloc = [&](size_t bytes) -> char* {
    char* q = p;
    p += (bytes + 255) & ~(size_t)255;
    return q;
  };
  float*    H        = (float*)alloc((size_t)n * 768 * 4);   // [gat|sage_r|q|k|v|skip]
  float*    concat   = (float*)alloc((size_t)n * 384 * 4);   // [gat|sage|tr]
  float*    gat_acc  = (float*)alloc((size_t)n * DIM * 4);
  float*    tr_acc   = (float*)alloc((size_t)n * DIM * 4);
  float*    nbr      = (float*)alloc((size_t)n * DIM * 4);
  float*    sage_tmp = (float*)alloc((size_t)n * DIM * 4);
  float*    pre      = (float*)alloc((size_t)n * DIM * 4);
  float*    a_src    = (float*)alloc((size_t)n * HEADS * 4);
  float*    a_dst    = (float*)alloc((size_t)n * HEADS * 4);
  float*    m_gat    = (float*)alloc((size_t)n * HEADS * 4);
  float*    s_gat    = (float*)alloc((size_t)n * HEADS * 4);
  float*    m_tr     = (float*)alloc((size_t)n * HEADS * 4);
  float*    s_tr     = (float*)alloc((size_t)n * HEADS * 4);
  float*    deg      = (float*)alloc((size_t)n * 4);
  _Float16* Bt6      = (_Float16*)alloc((size_t)768 * 128 * 2);
  float*    bias6    = (float*)alloc(768 * 4);
  _Float16* Btl      = (_Float16*)alloc((size_t)128 * 128 * 2);
  _Float16* Btf      = (_Float16*)alloc((size_t)128 * 384 * 2);
  (void)ws_size; (void)n_in; (void)out_size;

  // ---- pipeline
  k_init<<<(n * DIM + 255) / 256, 256, 0, stream>>>(m_gat, s_gat, m_tr, s_tr,
                                                    deg, gat_acc, tr_acc, nbr, n);
  k_prep<<<(768 * 128 + 255) / 256, 256, 0, stream>>>(gat_w, sage_wr, wq, wk, wv, wskip,
                                                      bq, bk, bv, bskip, sage_wl, fus_w,
                                                      Bt6, bias6, Btl, Btf);
  const int mtiles = (n + 15) / 16;
  k_wmma_gemm<128><<<mtiles, 128, 0, stream>>>(x, Bt6, bias6, H, n, 768, 768);

  k_alpha<<<(n * HEADS + 255) / 256, 256, 0, stream>>>(H, att_src, att_dst, a_src, a_dst, n);

  long long tE = E * HEADS;
  k_edge_max<<<(unsigned)((tE + 255) / 256), 256, 0, stream>>>(ei, E, H, a_src, a_dst,
                                                               m_gat, m_tr);
  k_self_max<<<(n * HEADS + 255) / 256, 256, 0, stream>>>(a_src, a_dst, m_gat, n);
  k_edge_acc<<<(unsigned)((tE + 255) / 256), 256, 0, stream>>>(ei, E, x, H, a_src, a_dst,
                                                               m_gat, m_tr, s_gat, s_tr,
                                                               gat_acc, tr_acc, nbr, deg);
  k_node_mid<<<(n * DIM + 255) / 256, 256, 0, stream>>>(H, a_src, a_dst, m_gat, s_gat,
                                                        s_tr, gat_acc, tr_acc, gat_b,
                                                        nbr, deg, concat, n);
  k_wmma_gemm<128><<<mtiles, 128, 0, stream>>>(nbr, Btl, sage_bl, sage_tmp, n, 128, 128);
  k_sage_norm<<<(n * 32 + 255) / 256, 256, 0, stream>>>(sage_tmp, H, concat, n);

  k_wmma_gemm<384><<<mtiles, 128, 0, stream>>>(concat, Btf, fus_b, pre, n, 128, 128);
  k_layernorm<<<(n * 32 + 255) / 256, 256, 0, stream>>>(pre, ln_g, ln_b, (float*)d_out, n);
}
